// LSTM_82429012345516
// MI455X (gfx1250) — matrix operand order repaired and checked by reading, not compiled, not run on hardware
//
#include <hip/hip_runtime.h>
#include <hip/hip_bf16.h>

#define SEQ_LEN 256
#define HIDDEN  2048
#define FOURH   8192
#define BATCH   128
#define CLASSES 10
#define NWG     128          // persistent workgroups; NWG * JPW == HIDDEN
#define JPW     16           // hidden units per workgroup
#define THREADS 128          // 4 waves: each owns a 32-row batch strip
#define COLS    (4 * JPW)    // 64 LDS-resident B columns per WG (4 gates x 16)
#define COLF4   257          // float4 stride per column: 4096B data + 16B pad
#define LDSBYTES (COLS * COLF4 * 16)   // 263,168 B < 320 KB WGP LDS

typedef __attribute__((ext_vector_type(16))) __bf16 v16bf;
typedef __attribute__((ext_vector_type(8)))  float  v8f;

union Frag { v16bf v; float4 f4[2]; };

__device__ __forceinline__ float fsig(float x)  { return 1.0f / (1.0f + __expf(-x)); }
__device__ __forceinline__ float ftanh(float x) { return 1.0f - 2.0f / (__expf(2.0f * x) + 1.0f); }

// Zero h0 broadcast buffer + grid-barrier counter (fresh every launch so
// graph replays are deterministic).
__global__ void lstm_init_kernel(unsigned* __restrict__ zwords, int n, unsigned* __restrict__ cnt) {
    int i = blockIdx.x * blockDim.x + threadIdx.x;
    if (i < n) zwords[i] = 0u;
    if (i == 0) *cnt = 0u;
}

// Pack Wh [K=2048, N=8192] fp32 row-major -> WhT [N=8192][K=2048] bf16
// (K-contiguous per column: a per-lane B fragment is two 16B runs of 8 k each).
__global__ void lstm_pack_wh(const float* __restrict__ Wh, __bf16* __restrict__ WhT) {
    int i = blockIdx.x * blockDim.x + threadIdx.x;   // 0 .. 8192*2048-1
    int n = i >> 11;
    int k = i & 2047;
    WhT[(size_t)n * HIDDEN + k] = (__bf16)Wh[(size_t)k * FOURH + n];
}

__global__ void __launch_bounds__(THREADS)
lstm_persistent_kernel(const float* __restrict__ x, const float* __restrict__ Wx,
                       const float* __restrict__ bias, const __bf16* __restrict__ WhT,
                       __bf16* __restrict__ hbuf, unsigned* __restrict__ cnt)
{
    extern __shared__ char smem[];
    const float4* smemB = (const float4*)smem;

    const int tid  = threadIdx.x;
    const int lane = tid & 31;
    const int wid  = tid >> 5;        // 0..3
    const int m0   = wid << 5;        // 32-row batch strip
    const int nlo  = lane & 15;       // N within tile / A-row selector
    const int hi16 = lane >> 4;       // B K-half selector
    const int hi8  = hi16 << 3;       // A K-half / C M-half selector
    const int j0   = blockIdx.x * JPW;

    // ---- One-time stage of this WG's 64 B columns (full K) into LDS via the
    //      CDNA5 async Global->LDS path (tracked by ASYNCcnt). ----
    {
        const unsigned ldsbase = (unsigned)(size_t)smem;
        const unsigned long long gbase = (unsigned long long)(size_t)WhT;
        for (int r = tid; r < COLS * 256; r += THREADS) {   // 16384 x 16B chunks
            int c    = r >> 8;                               // LDS column 0..63
            int sub  = r & 255;                              // 16B sub-chunk
            int gcol = (c >> 4) * HIDDEN + j0 + (c & 15);    // gate*H + j0 + n
            int      voff = gcol * (HIDDEN * 2) + sub * 16;  // byte offset vs WhT
            unsigned loff = ldsbase + (unsigned)(c * (COLF4 * 16) + sub * 16);
            asm volatile("global_load_async_to_lds_b128 %0, %1, %2"
                         :: "v"(loff), "v"(voff), "s"(gbase) : "memory");
        }
        asm volatile("s_wait_asynccnt 0x0" ::: "memory");
        __syncthreads();
    }

    // Hoisted per-column constants (INPUT_DIM == 1 -> Wx is one row).
    float wxv[4], bv[4];
#pragma unroll
    for (int g2 = 0; g2 < 4; ++g2) {        // gate order: g, i, f, o
        int col = g2 * HIDDEN + j0 + nlo;
        wxv[g2] = Wx[col];
        bv[g2]  = bias[col];
    }

    v8f cst[2] = {};                        // cell state, VGPR-resident all 255 steps

    __bf16* hb0 = hbuf;
    __bf16* hb1 = hbuf + (size_t)BATCH * HIDDEN;
    const int arow = (m0 + nlo) * HIDDEN + hi8;

    for (int t = 0; t < SEQ_LEN - 1; ++t) {
        const __bf16* hprev = (t & 1) ? hb1 : hb0;
        __bf16*       hnext = (t & 1) ? hb0 : hb1;

        v8f acc[2][4] = {};

        for (int k0 = 0; k0 < HIDDEN; k0 += 32) {
            Frag a0, a1;                                      // A: 2 x (16x32 bf16)
            const __bf16* ap = hprev + arow + k0;
            a0.f4[0] = *(const float4*)(ap);
            a0.f4[1] = *(const float4*)(ap + 16);
            a1.f4[0] = *(const float4*)(ap + 16 * HIDDEN);
            a1.f4[1] = *(const float4*)(ap + 16 * HIDDEN + 16);

            const int boff = (k0 >> 3) + hi16;                // float4 idx in column
#pragma unroll
            for (int g2 = 0; g2 < 4; ++g2) {                  // B from LDS, reused x2
                const float4* colp = smemB + (g2 * 16 + nlo) * COLF4;
                Frag b;
                b.f4[0] = colp[boff];
                b.f4[1] = colp[boff + 2];
                acc[0][g2] = __builtin_amdgcn_wmma_f32_16x16x32_bf16(
                    false, a0.v, false, b.v, (short)0, acc[0][g2], false, false);
                acc[1][g2] = __builtin_amdgcn_wmma_f32_16x16x32_bf16(
                    false, a1.v, false, b.v, (short)0, acc[1][g2], false, false);
            }
        }

        // Gate nonlinearities + cell update; rank-1 x_t term folded in here.
#pragma unroll
        for (int mi = 0; mi < 2; ++mi) {
#pragma unroll
            for (int v = 0; v < 8; ++v) {
                int   row = m0 + (mi << 4) + hi8 + v;
                float xv  = x[row * SEQ_LEN + t];
                float zg  = acc[mi][0][v] + xv * wxv[0] + bv[0];
                float zi  = acc[mi][1][v] + xv * wxv[1] + bv[1];
                float zf  = acc[mi][2][v] + xv * wxv[2] + bv[2];
                float zo  = acc[mi][3][v] + xv * wxv[3] + bv[3];
                float g   = ftanh(zg);
                float c   = g * fsig(zi) + cst[mi][v] * fsig(zf);
                cst[mi][v] = c;
                hnext[(size_t)row * HIDDEN + j0 + nlo] = (__bf16)(ftanh(c) * fsig(zo));
            }
        }

        // Device-wide barrier: monotone counter, sleep-spin on the leader lane.
        __threadfence();
        __syncthreads();
        if (tid == 0) {
            __hip_atomic_fetch_add(cnt, 1u, __ATOMIC_ACQ_REL, __HIP_MEMORY_SCOPE_AGENT);
            const unsigned target = (unsigned)NWG * (unsigned)(t + 1);
            while (__hip_atomic_load(cnt, __ATOMIC_ACQUIRE, __HIP_MEMORY_SCOPE_AGENT) < target)
                __builtin_amdgcn_s_sleep(1);
        }
        __syncthreads();
    }
}

// logits = h @ Wph + bp, softmax over 10 classes. One block per batch row.
__global__ void __launch_bounds__(256)
lstm_head_kernel(const __bf16* __restrict__ hfin, const float* __restrict__ Wph,
                 const float* __restrict__ bp, float* __restrict__ out)
{
    const int b   = blockIdx.x;
    const int tid = threadIdx.x;
    __shared__ float sred[256];
    __shared__ float slog[CLASSES];

    float part[CLASSES];
#pragma unroll
    for (int c = 0; c < CLASSES; ++c) part[c] = 0.f;

    for (int k = tid; k < HIDDEN; k += 256) {
        float hv = (float)hfin[(size_t)b * HIDDEN + k];
#pragma unroll
        for (int c = 0; c < CLASSES; ++c) part[c] += hv * Wph[k * CLASSES + c];
    }

    for (int c = 0; c < CLASSES; ++c) {
        sred[tid] = part[c];
        __syncthreads();
        for (int s = 128; s > 0; s >>= 1) {
            if (tid < s) sred[tid] += sred[tid + s];
            __syncthreads();
        }
        if (tid == 0) slog[c] = sred[0] + bp[c];
        __syncthreads();
    }

    if (tid == 0) {
        float m = slog[0];
        for (int c = 1; c < CLASSES; ++c) m = fmaxf(m, slog[c]);
        float s = 0.f, e[CLASSES];
        for (int c = 0; c < CLASSES; ++c) { e[c] = __expf(slog[c] - m); s += e[c]; }
        float inv = 1.f / s;
        for (int c = 0; c < CLASSES; ++c) out[b * CLASSES + c] = e[c] * inv;
    }
}

extern "C" void kernel_launch(void* const* d_in, const int* in_sizes, int n_in,
                              void* d_out, int out_size, void* d_ws, size_t ws_size,
                              hipStream_t stream)
{
    (void)in_sizes; (void)n_in; (void)out_size; (void)ws_size;
    const float* x   = (const float*)d_in[0];
    const float* Wx  = (const float*)d_in[1];
    const float* Wh  = (const float*)d_in[2];
    const float* b   = (const float*)d_in[3];
    const float* Wph = (const float*)d_in[4];
    const float* bp  = (const float*)d_in[5];
    float* out = (float*)d_out;

    // Workspace layout: [WhT bf16 32MB][hbuf double bf16 1MB][barrier counter]
    char* ws = (char*)d_ws;
    __bf16*   WhT  = (__bf16*)ws;
    __bf16*   hbuf = (__bf16*)(ws + (size_t)FOURH * HIDDEN * 2);
    unsigned* cnt  = (unsigned*)(ws + (size_t)FOURH * HIDDEN * 2
                                    + (size_t)2 * BATCH * HIDDEN * 2);

    const int zwords = (BATCH * HIDDEN * 2) / 4;   // zero hbuf[0]
    lstm_init_kernel<<<(zwords + 255) / 256, 256, 0, stream>>>((unsigned*)hbuf, zwords, cnt);
    lstm_pack_wh<<<(FOURH * HIDDEN) / 256, 256, 0, stream>>>(Wh, WhT);
    lstm_persistent_kernel<<<NWG, THREADS, LDSBYTES, stream>>>(x, Wx, b, WhT, hbuf, cnt);
    // 255 steps: last write (t = 254, even) lands in hbuf[1].
    lstm_head_kernel<<<BATCH, 256, 0, stream>>>(hbuf + (size_t)BATCH * HIDDEN, Wph, bp, out);
}
